// TopKFrozenEmbeddings_57801669869623
// MI455X (gfx1250) — matrix operand order repaired and checked by reading, not compiled, not run on hardware
//
#include <hip/hip_runtime.h>
#include <hip/hip_bf16.h>

// ---------------------------------------------------------------------------
// TopK frozen embeddings for MI455X (gfx1250, wave32, WMMA bf16 path)
//   inputs:      [4,2048,768] f32   -> N=8192 rows
//   embeddings:  [50257,768]  f32
//   out: probs f32 [8192], indices (as f32 values) [8192]
// ---------------------------------------------------------------------------

typedef __attribute__((ext_vector_type(16))) __bf16 v16bf;
typedef __attribute__((ext_vector_type(8)))  __bf16 v8bf;
typedef __attribute__((ext_vector_type(8)))  float  v8f;

#define N_ROWS   8192
#define D_FULL   768
#define D_RED    96           // 768 / 8
#define VOCAB    50257
#define VPAD     50272        // next multiple of 16 (3142 tiles)
#define NTILES   (VPAD / 16)
#define WAVES    8
#define NEG_INF  (-3.402823466e38f)

// ---------------------------------------------------------------------------
// Kernel 1: fold last dim by 8 and convert to bf16 (x_red and emb_red).
// ---------------------------------------------------------------------------
__global__ __launch_bounds__(256) void reduce_kernel(
    const float* __restrict__ x, const float* __restrict__ emb,
    __bf16* __restrict__ xr, __bf16* __restrict__ er)
{
    const long NX = (long)N_ROWS * D_RED;
    const long NE = (long)VPAD   * D_RED;
    long tid = (long)blockIdx.x * blockDim.x + threadIdx.x;
    if (tid < NX) {
        int row = (int)(tid / D_RED);
        int c   = (int)(tid % D_RED);
        const float* p = x + (long)row * D_FULL + c * 8;
        float s = 0.f;
        #pragma unroll
        for (int i = 0; i < 8; ++i) s += p[i];
        xr[tid] = (__bf16)s;
    } else if (tid < NX + NE) {
        long t  = tid - NX;
        int row = (int)(t / D_RED);
        int c   = (int)(t % D_RED);
        float s = 0.f;
        if (row < VOCAB) {
            const float* p = emb + (long)row * D_FULL + c * 8;
            #pragma unroll
            for (int i = 0; i < 8; ++i) s += p[i];
        }
        er[t] = (__bf16)s;
    }
}

// ---------------------------------------------------------------------------
// Top-5 bubble insert (rare path, exec-guarded by caller)
// ---------------------------------------------------------------------------
__device__ __forceinline__ void topk_insert(float (&s)[5], int (&ix)[5],
                                            float cs, int ci)
{
    #pragma unroll
    for (int j = 0; j < 5; ++j) {
        bool gt = cs > s[j];
        float os = s[j]; int oi = ix[j];
        s[j] = gt ? cs : os;  ix[j] = gt ? ci : oi;
        cs   = gt ? os : cs;  ci   = gt ? oi : ci;
    }
}

__device__ __forceinline__ void select_tile(const v8f& c, int col,
                                            float (&ts)[8][5], int (&ti)[8][5])
{
    if (col < VOCAB) {
        #pragma unroll
        for (int r = 0; r < 8; ++r) {
            float s = c[r];
            if (s > ts[r][4]) topk_insert(ts[r], ti[r], s, col);
        }
    }
}

// ---------------------------------------------------------------------------
// Kernel 2: fused approx-GEMM (v_wmma_f32_16x16x32_bf16) + streaming top-5
// + exact f32 rescore + softmax max / argmax.
// One workgroup (8 wave32s) per 16-row tile; vocab split across waves.
// Vocab loop unrolled x2 with independent C chains (hides WMMA D->src hazard).
// ---------------------------------------------------------------------------
__global__ __launch_bounds__(256, 1) void topk_kernel(
    const float* __restrict__ x, const float* __restrict__ emb,
    const __bf16* __restrict__ xr, const __bf16* __restrict__ er,
    float* __restrict__ out_probs, float* __restrict__ out_idx)
{
    // candidate pools: 16 rows x (8 waves * 16 lanes * 5) entries
    __shared__ float s_sc[16 * WAVES * 16 * 5];   // 40 KB
    __shared__ int   s_ix[16 * WAVES * 16 * 5];   // 40 KB
    __shared__ int   s_top[16 * 5];
    __shared__ float s_log[16 * 5];

    const int wave = threadIdx.x >> 5;
    const int lane = threadIdx.x & 31;
    const int m0   = blockIdx.x * 16;             // first of 16 query rows

    // ---- load A tiles once: x_red rows m0..m0+15, K = 0..95 (3 k-blocks)
    // A 16x32, this code's contiguous k order: lane<16 -> M=lane, K base 0; lane>=16 -> K base 16
    // regs 0-3 hold K=base..base+7, regs 4-7 hold K=base+8..base+15
    const int arow   = m0 + (lane & 15);
    const int akbase = (lane < 16) ? 0 : 16;
    union V16U { v16bf v; struct { v8bf lo; v8bf hi; } p; };
    v16bf a[3];
    #pragma unroll
    for (int kb = 0; kb < 3; ++kb) {
        const __bf16* ap = xr + (long)arow * D_RED + kb * 32 + akbase;
        V16U ua;
        ua.p.lo = *(const v8bf*)(ap);        // K = base .. base+7
        ua.p.hi = *(const v8bf*)(ap + 8);    // K = base+8 .. base+15
        a[kb] = ua.v;
    }

    // ---- per-lane running top-5 for the 8 C-rows this lane holds
    float ts[8][5];
    int   ti[8][5];
    #pragma unroll
    for (int r = 0; r < 8; ++r)
        #pragma unroll
        for (int j = 0; j < 5; ++j) { ts[r][j] = NEG_INF; ti[r][j] = 0; }

    // B 32x16, same contiguous k order as A: lane holds column N=lane%16; lane<16 -> K 0..15,
    // lane>=16 -> K 16..31 (contiguous 16 bf16 = one 32B chunk per k-block)
    const int bcol  = lane & 15;
    const int bkoff = (lane < 16) ? 0 : 16;

    int t = wave;
    for (; t + WAVES < NTILES; t += 2 * WAVES) {
        const int n0a = t * 16;
        const int n0b = (t + WAVES) * 16;
        const __bf16* bpa = er + (long)(n0a + bcol) * D_RED + bkoff;
        const __bf16* bpb = er + (long)(n0b + bcol) * D_RED + bkoff;
        // issue all 6 32B loads up front
        v16bf b00 = *(const v16bf*)(bpa);
        v16bf b01 = *(const v16bf*)(bpa + 32);
        v16bf b02 = *(const v16bf*)(bpa + 64);
        v16bf b10 = *(const v16bf*)(bpb);
        v16bf b11 = *(const v16bf*)(bpb + 32);
        v16bf b12 = *(const v16bf*)(bpb + 64);
        v8f c0 = {}, c1 = {};
        // two independent accumulation chains, interleaved
        c0 = __builtin_amdgcn_wmma_f32_16x16x32_bf16(false, a[0], false, b00, (short)0, c0, false, false);
        c1 = __builtin_amdgcn_wmma_f32_16x16x32_bf16(false, a[0], false, b10, (short)0, c1, false, false);
        c0 = __builtin_amdgcn_wmma_f32_16x16x32_bf16(false, a[1], false, b01, (short)0, c0, false, false);
        c1 = __builtin_amdgcn_wmma_f32_16x16x32_bf16(false, a[1], false, b11, (short)0, c1, false, false);
        c0 = __builtin_amdgcn_wmma_f32_16x16x32_bf16(false, a[2], false, b02, (short)0, c0, false, false);
        c1 = __builtin_amdgcn_wmma_f32_16x16x32_bf16(false, a[2], false, b12, (short)0, c1, false, false);
        select_tile(c0, n0a + bcol, ts, ti);
        select_tile(c1, n0b + bcol, ts, ti);
    }
    if (t < NTILES) {                      // tail: at most one tile per wave
        const int n0 = t * 16;
        const __bf16* bp = er + (long)(n0 + bcol) * D_RED + bkoff;
        v8f c = {};
        #pragma unroll
        for (int kb = 0; kb < 3; ++kb) {
            v16bf b = *(const v16bf*)(bp + kb * 32);
            c = __builtin_amdgcn_wmma_f32_16x16x32_bf16(
                    false, a[kb], false, b, (short)0, c, false, false);
        }
        select_tile(c, n0 + bcol, ts, ti);
    }

    // ---- dump per-lane candidates to LDS (row-major pools)
    #pragma unroll
    for (int r = 0; r < 8; ++r) {
        const int row  = r + ((lane < 16) ? 0 : 8);
        const int base = (row * (WAVES * 16) + wave * 16 + bcol) * 5;
        #pragma unroll
        for (int j = 0; j < 5; ++j) { s_sc[base + j] = ts[r][j]; s_ix[base + j] = ti[r][j]; }
    }
    __syncthreads();

    // ---- merge: 16 lanes of wave 0, one row each, scan 640 candidates
    if (threadIdx.x < 16) {
        const int row = threadIdx.x;
        float bs[5]; int bi[5];
        #pragma unroll
        for (int j = 0; j < 5; ++j) { bs[j] = NEG_INF; bi[j] = 0; }
        const int base = row * (WAVES * 16 * 5);
        for (int e = 0; e < WAVES * 16 * 5; ++e) {
            float s = s_sc[base + e];
            if (s > bs[4]) topk_insert(bs, bi, s, s_ix[base + e]);
        }
        #pragma unroll
        for (int j = 0; j < 5; ++j) s_top[row * 5 + j] = bi[j];
    }
    __syncthreads();

    // ---- exact rescore: 80 dot products of length 768, one wave per dot
    for (int j = wave; j < 16 * 5; j += WAVES) {
        const int row = j / 5, k = j % 5;
        const int idx = s_top[row * 5 + k];
        const float* xp = x   + (long)(m0 + row) * D_FULL;
        const float* ep = emb + (long)idx        * D_FULL;
        float acc = 0.f;
        for (int i = lane; i < D_FULL; i += 32) acc += xp[i] * ep[i];
        #pragma unroll
        for (int off = 16; off > 0; off >>= 1) acc += __shfl_down(acc, off, 32);
        if (lane == 0) s_log[row * 5 + k] = acc;
    }
    __syncthreads();

    // ---- softmax max prob + argmax index (argmax ties -> first, like jnp)
    if (threadIdx.x < 16) {
        const int row = threadIdx.x;
        float l[5];
        #pragma unroll
        for (int j = 0; j < 5; ++j) l[j] = s_log[row * 5 + j];
        float m = l[0]; int best = 0;
        #pragma unroll
        for (int j = 1; j < 5; ++j) if (l[j] > m) { m = l[j]; best = j; }
        float sum = 0.f;
        #pragma unroll
        for (int j = 0; j < 5; ++j) sum += __expf(l[j] - m);
        const int gr = m0 + row;
        out_probs[gr] = 1.0f / sum;                       // max of softmax
        out_idx[gr]   = (float)s_top[row * 5 + best];     // chosen vocab id
    }
}

// ---------------------------------------------------------------------------
extern "C" void kernel_launch(void* const* d_in, const int* in_sizes, int n_in,
                              void* d_out, int out_size, void* d_ws, size_t ws_size,
                              hipStream_t stream)
{
    const float* x   = (const float*)d_in[0];   // [4,2048,768]
    const float* emb = (const float*)d_in[1];   // [50257,768]

    __bf16* xr = (__bf16*)d_ws;                          // [8192][96]
    __bf16* er = xr + (size_t)N_ROWS * D_RED;            // [50272][96]

    const long total_red = (long)N_ROWS * D_RED + (long)VPAD * D_RED;
    const int  rblocks   = (int)((total_red + 255) / 256);
    reduce_kernel<<<rblocks, 256, 0, stream>>>(x, emb, xr, er);

    float* probs = (float*)d_out;
    float* idxs  = (float*)d_out + N_ROWS;
    topk_kernel<<<N_ROWS / 16, 256, 0, stream>>>(x, emb, xr, er, probs, idxs);
}
